// CausalSelfAttention_86165633893139
// MI455X (gfx1250) — matrix operand order repaired and checked by reading, not compiled, not run on hardware
//
#include <hip/hip_runtime.h>

// ---------------------------------------------------------------------------
// Causal self-attention for MI455X (gfx1250, wave32, WMMA bf16).
// Pipeline: cast->bf16, QKV GEMM (WMMA), flash attention (WMMA), out GEMM.
// Tile staging uses CDNA5 async global->LDS DMA (ASYNCcnt) with double
// buffering; all WMMA fragments for a K-step are materialized before the
// WMMA chain so LDS loads overlap the matrix pipe instead of draining
// dscnt before every wmma.
// ---------------------------------------------------------------------------

typedef __attribute__((ext_vector_type(16))) __bf16 bf16x16;
typedef __attribute__((ext_vector_type(8)))  float  f32x8;

static constexpr int B_ = 4, T_ = 2048, C_ = 1024, H_ = 16, D_ = 64;
static constexpr int M_ = B_ * T_;        // 8192 rows of x
static constexpr int NQKV_ = 3 * C_;      // 3072
static constexpr float QSCALE_ = 0.125f;  // 1/sqrt(64)

// ---------------- CDNA5 async global->LDS helpers --------------------------

__device__ __forceinline__ unsigned lds_off(const void* p) {
  // generic (flat) shared pointer: low 32 bits are the LDS byte offset
  return (unsigned)(uintptr_t)p;
}

__device__ __forceinline__ void async_load_b128(unsigned lds_byte_off,
                                                const void* gptr) {
  // GLOBAL_LOAD_ASYNC_TO_LDS_B128: VDST = LDS byte address VGPR,
  // VADDR = 64-bit global address. Tracked by ASYNCcnt.
  asm volatile("global_load_async_to_lds_b128 %0, %1, off"
               :
               : "v"(lds_byte_off), "v"(gptr)
               : "memory");
}

__device__ __forceinline__ void wait_async0() {
  asm volatile("s_wait_asynccnt 0" ::: "memory");
}

// ---------------- WMMA fragment helpers (wave32; contiguous k = 16*(lane/16) + elem, same for A and B) -----

// A fragment: 16x32 bf16, row-major source with given element stride.
// lanes 0-15: row = lane,    K in {0..15}
// lanes16-31: row = lane-16, K in {16..31}
__device__ __forceinline__ bf16x16 load_a_frag(const __bf16* base, int stride) {
  const int lane  = threadIdx.x & 31;
  const int row   = lane & 15;
  const int khalf = (lane >> 4) << 4;  // 0 or 16
  union { bf16x16 v; __bf16 e[16]; } u;
  const __bf16* p = base + row * stride + khalf;
#pragma unroll
  for (int i = 0; i < 8; ++i) {
    u.e[2 * i]     = p[2 * i];
    u.e[2 * i + 1] = p[2 * i + 1];
  }
  return u.v;
}

// B fragment: 32x16 bf16 (K x N), element (k,n) at base[k*stride + n].
// lanes 0-15: n = lane, K = 0..15 ; lanes 16-31: n = lane-16, K = 16..31
__device__ __forceinline__ bf16x16 load_b_frag(const __bf16* base, int stride) {
  const int lane = threadIdx.x & 31;
  const int n    = lane & 15;
  const int k0   = (lane >> 4) << 4;  // 0 or 16
  union { bf16x16 v; __bf16 e[16]; } u;
#pragma unroll
  for (int i = 0; i < 8; ++i) {
    u.e[2 * i]     = base[(k0 + 2 * i) * stride + n];
    u.e[2 * i + 1] = base[(k0 + 2 * i + 1) * stride + n];
  }
  return u.v;
}

// B fragment from transposed storage: element (k,n) at base[n*stride + k]
// (used for Q @ K^T where K is stored [keys][d]).
__device__ __forceinline__ bf16x16 load_bt_frag(const __bf16* base, int stride) {
  const int lane = threadIdx.x & 31;
  const int n    = lane & 15;
  const int k0   = (lane >> 4) << 4;
  union { bf16x16 v; __bf16 e[16]; } u;
  const __bf16* p = base + n * stride + k0;
#pragma unroll
  for (int i = 0; i < 8; ++i) {
    u.e[2 * i]     = p[2 * i];
    u.e[2 * i + 1] = p[2 * i + 1];
  }
  return u.v;
}

__device__ __forceinline__ f32x8 wmma_bf16(bf16x16 a, bf16x16 b, f32x8 c) {
  return __builtin_amdgcn_wmma_f32_16x16x32_bf16(false, a, false, b,
                                                 (short)0, c, false, false);
}

// ---------------- cast kernel ----------------------------------------------

__global__ void cast_f32_bf16(const float* __restrict__ in,
                              __bf16* __restrict__ out, int n) {
  int i = blockIdx.x * blockDim.x + threadIdx.x;
  int stride = gridDim.x * blockDim.x;
  for (; i < n; i += stride) out[i] = (__bf16)in[i];
}

// ---------------- GEMM: 128x64 block tile, BK=32, 8 waves, double-buffered -
// MODE 0: qkv = xb @ wqkvb + b_qkv, scatter into q(scaled)/k/v [B,H,T,D] bf16
// MODE 1: out = yb @ woutb + b_out, fp32 row-major [M,N]

template <int MODE>
__global__ __launch_bounds__(256) void gemm_bf16(
    const __bf16* __restrict__ A, const __bf16* __restrict__ Bm,
    const float* __restrict__ bias, int N, int K, float* __restrict__ outf,
    __bf16* __restrict__ qb, __bf16* __restrict__ kb, __bf16* __restrict__ vb) {
  __shared__ __bf16 As[2][128 * 40];  // 128 x 32, stride 40 (pad)
  __shared__ __bf16 Bs[2][32 * 72];   // 32 x 64, stride 72 (pad)

  const int tid  = threadIdx.x;
  const int wave = tid >> 5;
  const int lane = tid & 31;
  const int wm = wave & 1;   // M half: 0/1 -> 64 rows each
  const int wn = wave >> 1;  // N tile: 0..3 -> 16 cols each
  const int m0 = blockIdx.x * 128;
  const int n0 = blockIdx.y * 64;

  const unsigned as_off[2] = {lds_off(&As[0][0]), lds_off(&As[1][0])};
  const unsigned bs_off[2] = {lds_off(&Bs[0][0]), lds_off(&Bs[1][0])};

  // async-issue one K-step's tiles into LDS buffer `buf`
  auto issue_tiles = [&](int k0, int buf) {
    // A tile 128x32: 512 x b128 chunks, 2 per thread
#pragma unroll
    for (int c = 0; c < 2; ++c) {
      int cc = tid + c * 256;
      int row = cc >> 2, col8 = (cc & 3) * 8;
      async_load_b128(as_off[buf] + (unsigned)(row * 40 + col8) * 2,
                      A + (size_t)(m0 + row) * K + k0 + col8);
    }
    // B tile 32x64: 256 x b128 chunks, 1 per thread
    {
      int row = tid >> 3, col8 = (tid & 7) * 8;
      async_load_b128(bs_off[buf] + (unsigned)(row * 72 + col8) * 2,
                      Bm + (size_t)(k0 + row) * N + n0 + col8);
    }
  };

  f32x8 acc[4];
#pragma unroll
  for (int t = 0; t < 4; ++t)
#pragma unroll
    for (int i = 0; i < 8; ++i) acc[t][i] = 0.f;

  issue_tiles(0, 0);
  wait_async0();
  __syncthreads();

  int cur = 0;
  for (int k0 = 0; k0 < K; k0 += 32) {
    if (k0 + 32 < K) issue_tiles(k0 + 32, cur ^ 1);  // prefetch next tile

    // materialize all fragments first so LDS loads overlap the WMMA chain
    bf16x16 bfrag = load_b_frag(&Bs[cur][wn * 16], 72);
    bf16x16 afrag[4];
#pragma unroll
    for (int t = 0; t < 4; ++t)
      afrag[t] = load_a_frag(&As[cur][(wm * 64 + t * 16) * 40], 40);
#pragma unroll
    for (int t = 0; t < 4; ++t) acc[t] = wmma_bf16(afrag[t], bfrag, acc[t]);

    wait_async0();
    __syncthreads();
    cur ^= 1;
  }

  // epilogue: C/D tile layout -> row = vgpr + half*8, col = lane&15
  const int coln   = n0 + wn * 16 + (lane & 15);
  const int rowadd = (lane >> 4) << 3;
  const float bv = bias[coln];
#pragma unroll
  for (int t = 0; t < 4; ++t) {
#pragma unroll
    for (int i = 0; i < 8; ++i) {
      int m = m0 + wm * 64 + t * 16 + i + rowadd;
      float v = acc[t][i] + bv;
      if (MODE == 0) {
        int which = coln >> 10;  // 0=q 1=k 2=v
        int cc = coln & (C_ - 1);
        int h = cc >> 6, d = cc & 63;
        int bb = m >> 11, tt = m & (T_ - 1);
        size_t idx = ((size_t)(bb * H_ + h) * T_ + tt) * D_ + d;
        if (which == 0)      qb[idx] = (__bf16)(v * QSCALE_);
        else if (which == 1) kb[idx] = (__bf16)v;
        else                 vb[idx] = (__bf16)v;
      } else {
        outf[(size_t)m * N + coln] = v;
      }
    }
  }
}

// ---------------- flash attention ------------------------------------------
// grid: (T/128, B*H); block 256 threads = 8 waves; wave owns 16 query rows.
// K/V tiles (32 keys x 64) double-buffered through async global->LDS DMA.

__global__ __launch_bounds__(256) void attn_kernel(
    const __bf16* __restrict__ q, const __bf16* __restrict__ kmat,
    const __bf16* __restrict__ vmat, __bf16* __restrict__ y) {
  __shared__ __bf16 Klds[2][32 * 72];     // 32 keys x 64 d (pad 72)
  __shared__ __bf16 Vlds[2][32 * 72];
  __shared__ __bf16 Plds[8 * 16 * 40];    // per-wave 16x32 P staging (pad 40)

  const int tid  = threadIdx.x;
  const int wave = tid >> 5;
  const int lane = tid & 31;
  const int bh   = blockIdx.y;  // b*H + h
  const int b    = bh / H_;
  const int h    = bh % H_;
  const int qbase = blockIdx.x * 128;
  const int qrow0 = qbase + wave * 16;
  const int jend  = qbase + 128;  // causal: keys < jend

  const unsigned k_off[2] = {lds_off(&Klds[0][0]), lds_off(&Klds[1][0])};
  const unsigned v_off[2] = {lds_off(&Vlds[0][0]), lds_off(&Vlds[1][0])};

  // one b128 chunk per thread per tensor (32x64 bf16 = 4096B = 256 chunks)
  auto issue_kv = [&](int j, int buf) {
    int row = tid >> 3, col8 = (tid & 7) * 8;
    unsigned loff = (unsigned)(row * 72 + col8) * 2;
    const __bf16* kg = kmat + ((size_t)bh * T_ + j) * D_;
    const __bf16* vg = vmat + ((size_t)bh * T_ + j) * D_;
    async_load_b128(k_off[buf] + loff, kg + row * 64 + col8);
    async_load_b128(v_off[buf] + loff, vg + row * 64 + col8);
  };

  // Q rows for this wave, as two 16x32 A fragments (q pre-scaled by 1/sqrt(D))
  const __bf16* qptr = q + ((size_t)bh * T_ + qrow0) * D_;
  bf16x16 qa0 = load_a_frag(qptr, D_);
  bf16x16 qa1 = load_a_frag(qptr + 32, D_);

  f32x8 o[4];
#pragma unroll
  for (int t = 0; t < 4; ++t)
#pragma unroll
    for (int i = 0; i < 8; ++i) o[t][i] = 0.f;
  float mrow[8], lrow[8];
#pragma unroll
  for (int i = 0; i < 8; ++i) { mrow[i] = -__builtin_inff(); lrow[i] = 0.f; }

  const int rowadd = (lane >> 4) << 3;
  const int coln   = lane & 15;

  issue_kv(0, 0);
  wait_async0();
  __syncthreads();

  int cur = 0;
  for (int j = 0; j < jend; j += 32) {
    if (j + 32 < jend) issue_kv(j + 32, cur ^ 1);  // prefetch next KV block

    // S = Q @ K^T : two 16x16 tiles over 32 keys, contraction d=64 (2 steps)
    // materialize all K^T fragments, then run the WMMA chain
    const __bf16* Kc = &Klds[cur][0];
    bf16x16 kf00 = load_bt_frag(Kc + 0 * 72 + 0, 72);
    bf16x16 kf01 = load_bt_frag(Kc + 0 * 72 + 32, 72);
    bf16x16 kf10 = load_bt_frag(Kc + 16 * 72 + 0, 72);
    bf16x16 kf11 = load_bt_frag(Kc + 16 * 72 + 32, 72);
    f32x8 s0, s1;
#pragma unroll
    for (int i = 0; i < 8; ++i) { s0[i] = 0.f; s1[i] = 0.f; }
    s0 = wmma_bf16(qa0, kf00, s0);
    s1 = wmma_bf16(qa0, kf10, s1);
    s0 = wmma_bf16(qa1, kf01, s0);
    s1 = wmma_bf16(qa1, kf11, s1);

    // online softmax (row stats reduced across the 16-lane half-wave)
    float p0a[8], p1a[8], alpha[8];
#pragma unroll
    for (int i = 0; i < 8; ++i) {
      int qg = qrow0 + i + rowadd;
      float a0 = ((j + coln) <= qg)      ? s0[i] : -__builtin_inff();
      float a1 = ((j + 16 + coln) <= qg) ? s1[i] : -__builtin_inff();
      float mx = fmaxf(a0, a1);
#pragma unroll
      for (int msk = 1; msk < 16; msk <<= 1)
        mx = fmaxf(mx, __shfl_xor(mx, msk, 16));
      float mnew = fmaxf(mrow[i], mx);
      float al = __expf(mrow[i] - mnew);
      float p0 = __expf(a0 - mnew);
      float p1 = __expf(a1 - mnew);
      float sum = p0 + p1;
#pragma unroll
      for (int msk = 1; msk < 16; msk <<= 1) sum += __shfl_xor(sum, msk, 16);
      lrow[i] = lrow[i] * al + sum;
      mrow[i] = mnew;
      alpha[i] = al;
      p0a[i] = p0;
      p1a[i] = p1;
    }
#pragma unroll
    for (int nt = 0; nt < 4; ++nt)
#pragma unroll
      for (int i = 0; i < 8; ++i) o[nt][i] *= alpha[i];

    // stage P (C-layout -> A-layout) through per-wave LDS
    __bf16* pst = Plds + wave * 16 * 40;
#pragma unroll
    for (int i = 0; i < 8; ++i) {
      int r = i + rowadd;
      pst[r * 40 + coln]      = (__bf16)p0a[i];
      pst[r * 40 + 16 + coln] = (__bf16)p1a[i];
    }
    asm volatile("s_wait_dscnt 0" ::: "memory");  // intra-wave LDS RAW
    bf16x16 pfrag = load_a_frag(pst, 40);

    // O += P @ V  (contraction over 32 keys, 4 N tiles of 16 over D=64)
    const __bf16* Vc = &Vlds[cur][0];
    bf16x16 vfrag[4];
#pragma unroll
    for (int nt = 0; nt < 4; ++nt) vfrag[nt] = load_b_frag(Vc + nt * 16, 72);
#pragma unroll
    for (int nt = 0; nt < 4; ++nt)
      o[nt] = wmma_bf16(pfrag, vfrag[nt], o[nt]);

    wait_async0();
    __syncthreads();
    cur ^= 1;
  }

  // normalize and write y [B,T,C] bf16 at column block h*64
  __bf16* yp = y + ((size_t)b * T_ + qrow0) * C_ + h * D_;
#pragma unroll
  for (int i = 0; i < 8; ++i) {
    float inv = 1.f / lrow[i];
    int r = i + rowadd;
#pragma unroll
    for (int nt = 0; nt < 4; ++nt)
      yp[(size_t)r * C_ + nt * 16 + coln] = (__bf16)(o[nt][i] * inv);
  }
}

// ---------------- host launch ----------------------------------------------

extern "C" void kernel_launch(void* const* d_in, const int* in_sizes, int n_in,
                              void* d_out, int out_size, void* d_ws,
                              size_t ws_size, hipStream_t stream) {
  const float* x    = (const float*)d_in[0];
  const float* wqkv = (const float*)d_in[1];
  const float* bqkv = (const float*)d_in[2];
  const float* wout = (const float*)d_in[3];
  const float* bout = (const float*)d_in[4];
  float* out = (float*)d_out;

  // workspace carve-out (~92 MB total)
  char* ws = (char*)d_ws;
  size_t off = 0;
  auto take = [&](size_t bytes) -> char* {
    char* p = ws + off;
    off += (bytes + 255) & ~(size_t)255;
    return p;
  };
  __bf16* xb    = (__bf16*)take((size_t)M_ * C_ * 2);
  __bf16* wqkvb = (__bf16*)take((size_t)C_ * NQKV_ * 2);
  __bf16* woutb = (__bf16*)take((size_t)C_ * C_ * 2);
  __bf16* qb    = (__bf16*)take((size_t)B_ * H_ * T_ * D_ * 2);
  __bf16* kb    = (__bf16*)take((size_t)B_ * H_ * T_ * D_ * 2);
  __bf16* vb    = (__bf16*)take((size_t)B_ * H_ * T_ * D_ * 2);
  __bf16* yb    = (__bf16*)take((size_t)M_ * C_ * 2);

  cast_f32_bf16<<<1024, 256, 0, stream>>>(x, xb, M_ * C_);
  cast_f32_bf16<<<512, 256, 0, stream>>>(wqkv, wqkvb, C_ * NQKV_);
  cast_f32_bf16<<<256, 256, 0, stream>>>(wout, woutb, C_ * C_);

  gemm_bf16<0><<<dim3(M_ / 128, NQKV_ / 64), 256, 0, stream>>>(
      xb, wqkvb, bqkv, NQKV_, C_, nullptr, qb, kb, vb);

  attn_kernel<<<dim3(T_ / 128, B_ * H_), 256, 0, stream>>>(qb, kb, vb, yb);

  gemm_bf16<1><<<dim3(M_ / 128, C_ / 64), 256, 0, stream>>>(
      yb, woutb, bout, C_, C_, out, nullptr, nullptr, nullptr);
}
